// dual_attn_fusion_89644557402222
// MI455X (gfx1250) — matrix operand order repaired and checked by reading, not compiled, not run on hardware
//
#include <hip/hip_runtime.h>
#include <math.h>

// ---------------------------------------------------------------------------
// Problem constants (from reference): B=1024, N=512, D=48, H=128
// ---------------------------------------------------------------------------
#define BB 1024
#define NN 512
#define DD 48
#define HH 128

typedef __attribute__((ext_vector_type(16))) __bf16 bf16x16;
typedef __attribute__((ext_vector_type(8)))  float  v8f;

union Frag {
    bf16x16  v;
    unsigned u[8];
};

// f32 -> packed 2x bf16 (RNE). Prefer the single-instruction packer.
#if __has_builtin(__builtin_amdgcn_cvt_pk_bf16_f32)
typedef __attribute__((ext_vector_type(2))) __bf16 bf16x2;
__device__ __forceinline__ unsigned pack_bf16x2(float lo, float hi) {
    union { bf16x2 v; unsigned u; } x;
    x.v = __builtin_amdgcn_cvt_pk_bf16_f32(lo, hi);
    return x.u;
}
#else
__device__ __forceinline__ unsigned bf16_bits(float f) {
    union { float f; unsigned u; } x;
    x.f = f;
    unsigned u = x.u;
    return (u + 0x7fffu + ((u >> 16) & 1u)) >> 16;
}
__device__ __forceinline__ unsigned pack_bf16x2(float lo, float hi) {
    return bf16_bits(lo) | (bf16_bits(hi) << 16);
}
#endif

// hardware V_TANH_F32 when available (gfx1250 TRANS op), else ocml
__device__ __forceinline__ float fast_tanh(float x) {
#if __has_builtin(__builtin_amdgcn_tanhf)
    return __builtin_amdgcn_tanhf(x);
#elif __has_builtin(__builtin_amdgcn_tanh_f32)
    return __builtin_amdgcn_tanh_f32(x);
#else
    return tanhf(x);
#endif
}

// ---------------------------------------------------------------------------
// K1: cosine similarity sim[b,n] = <s,i>/max(|s||i|, eps). Thread per (b,n).
// ---------------------------------------------------------------------------
__global__ void sim_kernel(const float* __restrict__ img,
                           const float* __restrict__ spatial,
                           float* __restrict__ sim) {
    int idx = blockIdx.x * blockDim.x + threadIdx.x;
    if (idx >= BB * NN) return;
    const float4* sp = (const float4*)(spatial + (size_t)idx * DD);
    const float4* ip = (const float4*)(img     + (size_t)idx * DD);
    float dot = 0.f, ns = 0.f, ni = 0.f;
#pragma unroll
    for (int p = 0; p < DD / 4; ++p) {
        float4 s = sp[p];
        float4 i = ip[p];
        dot += s.x * i.x + s.y * i.y + s.z * i.z + s.w * i.w;
        ns  += s.x * s.x + s.y * s.y + s.z * s.z + s.w * s.w;
        ni  += i.x * i.x + i.y * i.y + i.z * i.z + i.w * i.w;
    }
    sim[idx] = dot / fmaxf(sqrtf(ns) * sqrtf(ni), 1e-8f);
}

// ---------------------------------------------------------------------------
// K2: per-batch MLP via bf16 WMMA + fused column-mean accumulation.
//   wsum[b][k] = sum_n ( tanh(z[b,k,n,:] @ W1 + b1) @ W2 )
//   gM[b][k][d] = sum_n z[b,k,n,d]        (exact fp32, from the staged tiles)
// Block = one batch b, 128 threads = 4 waves, each wave owns 16 row-tiles.
// No in-loop block barriers: each wave owns its LDS A-tile and per-wave DS
// ops are in-order (ISA 7.3), so waves run fully decoupled.
// ---------------------------------------------------------------------------
__global__ void __launch_bounds__(128)
mlp_wmma_kernel(const float* __restrict__ spatial,
                const float* __restrict__ temporal,
                const float* __restrict__ W1,
                const float* __restrict__ b1,
                const float* __restrict__ W2,
                const float* __restrict__ sim,
                float* __restrict__ wsum,
                float* __restrict__ gM) {
    // B fragments: [c(2)][cc(8)][j(8)][lane(32)] packed pairs -> 4096 u32 = 16 KB
    __shared__ unsigned sB[2 * 8 * 8 * 32];
    // per-wave A tile: 16 rows x 32 u32 (64 bf16) -> 8 KB total
    __shared__ unsigned sA[4][16 * 32];
    __shared__ float sb1[HH];
    __shared__ float sw2[HH];
    __shared__ float sred[4][2];
    // per-lane column partial sums for the deterministic mean reduction
    __shared__ float sAccS[4][32][3][4];   // 6 KB
    __shared__ float sAccT[4][32][3][4];   // 6 KB

    const int b    = blockIdx.x;
    const int tid  = threadIdx.x;
    const int lane = tid & 31;
    const int wave = tid >> 5;
    const int row  = lane & 15;   // M index (A) / N col-within-chunk (B,C)
    const int half = lane >> 4;

    if (tid < HH) {
        sb1[tid] = b1[tid];
        sw2[tid] = W2[tid];
    }
    // Pre-pack W1 -> bf16 B fragments (K padded to 64 with zeros).
    // B layout 32x16: lanes 0-15 K=0..15 (VGPR j: K=2j,2j+1), lanes 16-31
    // K=16..31; chunk c adds 32*c to K.
    for (int e = tid; e < 4096; e += 128) {
        int l  = e & 31;
        int j  = (e >> 5) & 7;
        int cc = (e >> 8) & 7;
        int c  = (e >> 11) & 1;
        int col = 16 * cc + (l & 15);
        int k0  = 32 * c + 16 * (l >> 4) + 2 * j;
        float f0 = (k0     < DD) ? W1[(size_t)k0 * HH + col]       : 0.f;
        float f1 = (k0 + 1 < DD) ? W1[(size_t)(k0 + 1) * HH + col] : 0.f;
        sB[e] = pack_bf16x2(f0, f1);
    }
    // Zero the K=48..63 pad region of this wave's A tile (written once).
    for (int i = lane; i < 16 * 8; i += 32) {
        int r = i >> 3, q = i & 7;
        sA[wave][r * 32 + 24 + q] = 0u;
    }
    __syncthreads();

    float sum0 = 0.f, sum1 = 0.f;        // tanh-MLP partial sums per branch
    float4 accS[3] = {};                 // sum_n sim*spatial, per d-quad slot
    float4 accT[3] = {};                 // sum_n temporal

    // Wave handles tiles t = wave + 4*i, i = 0..15  (t in 0..63):
    //   k_sel = t>>5 (0: sim-weighted spatial, 1: temporal), n0 = (t&31)*16
    for (int i = 0; i < 16; ++i) {
        const int t     = wave + 4 * i;
        const int k_sel = t >> 5;
        const int n0    = (t & 31) * 16;
        const float* src = (k_sel ? temporal : spatial)
                         + ((size_t)b * NN + n0) * DD;

        // Stage 16x48 fp32 tile -> bf16 LDS (scale spatial rows by sim),
        // and fold the scaled fp32 values into the column accumulators.
        // p = lane + 32m: d-quad dq = (lane%12 + 8m) % 12 -> slot = m % 3.
#pragma unroll
        for (int m = 0; m < 6; ++m) {
            int p  = lane + 32 * m;
            int r  = p / (DD / 4);
            int dq = p % (DD / 4);
            float4 v = ((const float4*)src)[p];
            if (!k_sel) {
                float s = sim[(size_t)b * NN + n0 + r];
                v.x *= s; v.y *= s; v.z *= s; v.w *= s;
            }
            const int slot = m % 3;
            if (k_sel) {
                accT[slot].x += v.x; accT[slot].y += v.y;
                accT[slot].z += v.z; accT[slot].w += v.w;
            } else {
                accS[slot].x += v.x; accS[slot].y += v.y;
                accS[slot].z += v.z; accS[slot].w += v.w;
            }
            sA[wave][r * 32 + dq * 2 + 0] = pack_bf16x2(v.x, v.y);
            sA[wave][r * 32 + dq * 2 + 1] = pack_bf16x2(v.z, v.w);
        }

        // A fragments (contiguous K order k = 16*half + s, matching B), one per K-chunk.
        // Same-wave DS ops are in-order: no barrier needed vs the stores.
        Frag a[2];
#pragma unroll
        for (int c = 0; c < 2; ++c) {
            int base = row * 32 + 16 * c + 8 * half;
#pragma unroll
            for (int j = 0; j < 4; ++j) {
                a[c].u[j]     = sA[wave][base + j];
                a[c].u[4 + j] = sA[wave][base + 4 + j];
            }
        }

        v8f acc[8] = {};
#pragma unroll
        for (int cc = 0; cc < 8; ++cc) {
#pragma unroll
            for (int c = 0; c < 2; ++c) {
                Frag bf;
#pragma unroll
                for (int j = 0; j < 8; ++j)
                    bf.u[j] = sB[((c * 8 + cc) * 8 + j) * 32 + lane];
                acc[cc] = __builtin_amdgcn_wmma_f32_16x16x32_bf16(
                    false, a[c].v, false, bf.v, (short)0, acc[cc],
                    false, false);
            }
        }

        // epilogue: + b1, tanh, dot with W2, full-tile sum
        float csum = 0.f;
#pragma unroll
        for (int cc = 0; cc < 8; ++cc) {
            int col  = 16 * cc + row;
            float bb = sb1[col];
            float ww = sw2[col];
#pragma unroll
            for (int r = 0; r < 8; ++r)
                csum += fast_tanh(acc[cc][r] + bb) * ww;
        }
        if (k_sel) sum1 += csum; else sum0 += csum;
    }

    // wave32 butterfly reduce of the MLP sums
    for (int off = 16; off; off >>= 1) {
        sum0 += __shfl_xor(sum0, off);
        sum1 += __shfl_xor(sum1, off);
    }
    if (lane == 0) {
        sred[wave][0] = sum0;
        sred[wave][1] = sum1;
    }
    // park per-lane column accumulators for the deterministic reduction
#pragma unroll
    for (int s = 0; s < 3; ++s) {
        sAccS[wave][lane][s][0] = accS[s].x;
        sAccS[wave][lane][s][1] = accS[s].y;
        sAccS[wave][lane][s][2] = accS[s].z;
        sAccS[wave][lane][s][3] = accS[s].w;
        sAccT[wave][lane][s][0] = accT[s].x;
        sAccT[wave][lane][s][1] = accT[s].y;
        sAccT[wave][lane][s][2] = accT[s].z;
        sAccT[wave][lane][s][3] = accT[s].w;
    }
    __syncthreads();

    if (tid == 0) {
        wsum[b * 2 + 0] = (sred[0][0] + sred[1][0]) + (sred[2][0] + sred[3][0]);
        wsum[b * 2 + 1] = (sred[0][1] + sred[1][1]) + (sred[2][1] + sred[3][1]);
    }
    // Fixed-order combine: thread d sums the ~8 (lane,slot) pairs per wave
    // whose d-quad matches.  slot0: lane%12==dq, slot1: lane%12==(dq+4)%12,
    // slot2: lane%12==(dq+8)%12.
    if (tid < DD) {
        const int dq = tid >> 2, comp = tid & 3;
        const int bases[3] = { dq, (dq + 4) % 12, (dq + 8) % 12 };
        float ts = 0.f, tt = 0.f;
#pragma unroll
        for (int w = 0; w < 4; ++w)
#pragma unroll
            for (int s = 0; s < 3; ++s)
#pragma unroll
                for (int rep = 0; rep < 3; ++rep) {
                    int l = bases[s] + 12 * rep;
                    if (l < 32) {
                        ts += sAccS[w][l][s][comp];
                        tt += sAccT[w][l][s][comp];
                    }
                }
        gM[((size_t)b * 2 + 0) * DD + tid] = ts;
        gM[((size_t)b * 2 + 1) * DD + tid] = tt;
    }
}

// ---------------------------------------------------------------------------
// K3: softmax over the 2 branch logits, combine the precomputed means.
// ---------------------------------------------------------------------------
__global__ void finalize_kernel(const float* __restrict__ gM,
                                const float* __restrict__ wsum,
                                float* __restrict__ out) {
    int b = blockIdx.x;
    int d = threadIdx.x;
    float w0 = wsum[b * 2 + 0] * (1.f / NN);
    float w1 = wsum[b * 2 + 1] * (1.f / NN);
    float m  = fmaxf(w0, w1);
    float e0 = expf(w0 - m), e1 = expf(w1 - m);
    float inv = 1.f / (e0 + e1);
    float beta0 = e0 * inv, beta1 = e1 * inv;
    if (d >= DD) return;
    float ms = gM[((size_t)b * 2 + 0) * DD + d];
    float mt = gM[((size_t)b * 2 + 1) * DD + d];
    out[b * DD + d] = (beta0 * ms + beta1 * mt) * (1.f / NN);
}

// ---------------------------------------------------------------------------
extern "C" void kernel_launch(void* const* d_in, const int* in_sizes, int n_in,
                              void* d_out, int out_size, void* d_ws, size_t ws_size,
                              hipStream_t stream) {
    const float* img      = (const float*)d_in[0];
    const float* spatial  = (const float*)d_in[1];
    const float* temporal = (const float*)d_in[2];
    const float* W1       = (const float*)d_in[3];
    const float* b1       = (const float*)d_in[4];
    const float* W2       = (const float*)d_in[5];
    float* out = (float*)d_out;

    float* sim  = (float*)d_ws;              // B*N floats (2 MB)
    float* wsum = sim + (size_t)BB * NN;     // B*2 floats
    float* gM   = wsum + (size_t)BB * 2;     // B*2*D floats (384 KB)

    sim_kernel<<<(BB * NN + 255) / 256, 256, 0, stream>>>(img, spatial, sim);
    mlp_wmma_kernel<<<BB, 128, 0, stream>>>(spatial, temporal, W1, b1, W2,
                                            sim, wsum, gM);
    finalize_kernel<<<BB, 64, 0, stream>>>(gM, wsum, out);
}
